// MultiheadSelfAttention_19696720020269
// MI455X (gfx1250) — matrix operand order repaired and checked by reading, not compiled, not run on hardware
//
#include <hip/hip_runtime.h>
#include <hip/hip_bf16.h>
#include <math.h>

#define EMBED 1024
#define NHEAD 16
#define HDIM  64
#define BATCH 4
#define SEQ   2048
#define TOK   (BATCH * SEQ)   // 8192 tokens

typedef __attribute__((ext_vector_type(16))) __bf16 bf16x16;
typedef __attribute__((ext_vector_type(8)))  float  f32x8;

// ---------- helpers ----------

static __device__ __forceinline__ unsigned short f2bf(float f) {
  // round-to-nearest-even fp32 -> bf16
  unsigned u = __builtin_bit_cast(unsigned, f);
  unsigned r = u + 0x7FFFu + ((u >> 16) & 1u);
  return (unsigned short)(r >> 16);
}

static __device__ __forceinline__ bf16x16 mk_frag(uint4 a, uint4 b) {
  union { unsigned u[8]; bf16x16 v; } x;
  x.u[0] = a.x; x.u[1] = a.y; x.u[2] = a.z; x.u[3] = a.w;
  x.u[4] = b.x; x.u[5] = b.y; x.u[6] = b.z; x.u[7] = b.w;
  return x.v;
}

static __device__ __forceinline__ f32x8 wmma_bf16(bf16x16 a, bf16x16 b, f32x8 c) {
  return __builtin_amdgcn_wmma_f32_16x16x32_bf16(false, a, false, b, (short)0, c,
                                                 false, false);
}

// Async global -> LDS copy, 16 bytes per lane (GLOBAL_LOAD_ASYNC_TO_LDS_B128,
// tracked by ASYNCcnt). LDS operand is the 32-bit LDS byte address (low 32 bits
// of the generic pointer to __shared__), global address is a 64-bit VGPR pair.
static __device__ __forceinline__ void g2l_b128(void* lds, const void* g) {
  unsigned l = (unsigned)(uintptr_t)lds;
  unsigned long long a = (unsigned long long)(uintptr_t)g;
  asm volatile("global_load_async_to_lds_b128 %0, %1, off" ::"v"(l), "v"(a)
               : "memory");
}

static __device__ __forceinline__ void wait_async0() {
  asm volatile("s_wait_asynccnt 0x0" ::: "memory");
}

// ---------- kernel 0a: x fp32 -> bf16 row-major ----------

__global__ void cvt_x(const float* __restrict__ x, unsigned short* __restrict__ xbf) {
  int i = blockIdx.x * 256 + threadIdx.x;          // 2,097,152 float4 total
  float4 v = ((const float4*)x)[i];
  unsigned lo = (unsigned)f2bf(v.x) | ((unsigned)f2bf(v.y) << 16);
  unsigned hi = (unsigned)f2bf(v.z) | ((unsigned)f2bf(v.w) << 16);
  ((uint2*)xbf)[i] = make_uint2(lo, hi);
}

// ---------- kernel 0b: weights fp32 [k][n] -> bf16 col-major [n][k] ----------

__global__ void cvt_w(const float* __restrict__ Wq, const float* __restrict__ Wk,
                      const float* __restrict__ Wv, const float* __restrict__ Wo,
                      unsigned short* __restrict__ wbf) {
  __shared__ float t[32][33];
  const int wsel = blockIdx.z;
  const float* W = (wsel == 0) ? Wq : (wsel == 1) ? Wk : (wsel == 2) ? Wv : Wo;
  unsigned short* dst = wbf + (size_t)wsel * EMBED * EMBED;
  const int kb = blockIdx.y * 32, nb = blockIdx.x * 32;
  const int tx = threadIdx.x, ty = threadIdx.y;    // block (32, 8)
#pragma unroll
  for (int i = 0; i < 32; i += 8)
    t[ty + i][tx] = W[(size_t)(kb + ty + i) * EMBED + nb + tx];
  __syncthreads();
#pragma unroll
  for (int i = 0; i < 32; i += 8)
    dst[(size_t)(nb + ty + i) * EMBED + kb + tx] = f2bf(t[tx][ty + i]);
}

// ---------- kernel 1: fused QKV projections (z = 0/1/2 -> Q/K/V) ----------
// C = Xbf[8192x1024] @ Wbf(col-major) + bias ; Q scaled by 1/8 ; V stored transposed.
// Double-buffered LDS tiles filled by async global->LDS DMA.

#define PITCH1 40   // LDS row pitch in ushort (80B: 16B-aligned, conflict-free)

__global__ __launch_bounds__(256, 1) void qkv_gemm(
    const unsigned short* __restrict__ xbf, const unsigned short* __restrict__ wbf,
    const float* __restrict__ bq, const float* __restrict__ bk,
    const float* __restrict__ bv,
    unsigned short* __restrict__ qout, unsigned short* __restrict__ kout,
    unsigned short* __restrict__ vtout) {
  __shared__ unsigned short lA[2][128 * PITCH1];
  __shared__ unsigned short lB[2][128 * PITCH1];

  const int wsel = blockIdx.z;
  const unsigned short* W = wbf + (size_t)wsel * EMBED * EMBED;
  const float* bias = (wsel == 0) ? bq : (wsel == 1) ? bk : bv;
  const int m0 = blockIdx.y * 128, n0 = blockIdx.x * 128;
  const int tid = threadIdx.x;
  const int wave = tid >> 5, lane = tid & 31, half = lane >> 4, l16 = lane & 15;
  const int wm = (wave & 1) * 64, wn = (wave >> 1) * 32;

  // this thread's two 16B staging chunks (128 rows x 4 chunks = 512 chunks)
  const int r0 = tid >> 2, o0 = (tid & 3) * 8;
  const int r1 = (tid + 256) >> 2, o1 = ((tid + 256) & 3) * 8;

  f32x8 acc[4][2];
#pragma unroll
  for (int i = 0; i < 4; i++)
#pragma unroll
    for (int j = 0; j < 2; j++)
#pragma unroll
      for (int r = 0; r < 8; r++) acc[i][j][r] = 0.0f;

  // prologue: async-stage tile k0 = 0 into buffer 0
  g2l_b128(&lA[0][r0 * PITCH1 + o0], &xbf[(size_t)(m0 + r0) * EMBED + o0]);
  g2l_b128(&lA[0][r1 * PITCH1 + o1], &xbf[(size_t)(m0 + r1) * EMBED + o1]);
  g2l_b128(&lB[0][r0 * PITCH1 + o0], &W[(size_t)(n0 + r0) * EMBED + o0]);
  g2l_b128(&lB[0][r1 * PITCH1 + o1], &W[(size_t)(n0 + r1) * EMBED + o1]);
  wait_async0();
  __syncthreads();

  int buf = 0;
  for (int k0 = 0; k0 < EMBED; k0 += 32) {
    const int nb = buf ^ 1;
    if (k0 + 32 < EMBED) {  // uniform: async-prefetch next K-tile into other buffer
      const int kk = k0 + 32;
      g2l_b128(&lA[nb][r0 * PITCH1 + o0], &xbf[(size_t)(m0 + r0) * EMBED + kk + o0]);
      g2l_b128(&lA[nb][r1 * PITCH1 + o1], &xbf[(size_t)(m0 + r1) * EMBED + kk + o1]);
      g2l_b128(&lB[nb][r0 * PITCH1 + o0], &W[(size_t)(n0 + r0) * EMBED + kk + o0]);
      g2l_b128(&lB[nb][r1 * PITCH1 + o1], &W[(size_t)(n0 + r1) * EMBED + kk + o1]);
    }

    bf16x16 bfr[2];
#pragma unroll
    for (int nt = 0; nt < 2; nt++) {
      const unsigned short* p = &lB[buf][(wn + nt * 16 + l16) * PITCH1 + half * 8];
      bfr[nt] = mk_frag(*(const uint4*)p, *(const uint4*)(p + 16));
    }
#pragma unroll
    for (int mt = 0; mt < 4; mt++) {
      const unsigned short* p = &lA[buf][(wm + mt * 16 + l16) * PITCH1 + half * 8];
      bf16x16 af = mk_frag(*(const uint4*)p, *(const uint4*)(p + 16));
#pragma unroll
      for (int nt = 0; nt < 2; nt++) acc[mt][nt] = wmma_bf16(af, bfr[nt], acc[mt][nt]);
    }

    wait_async0();     // own async copies into the other buffer are complete
    __syncthreads();   // everyone's are; safe to compute from it / reuse this one
    buf = nb;
  }

  // epilogue: bias, scale (Q), head-major scatter (V transposed [b,h,d,s])
#pragma unroll
  for (int mt = 0; mt < 4; mt++)
#pragma unroll
    for (int nt = 0; nt < 2; nt++)
#pragma unroll
      for (int r = 0; r < 8; r++) {
        int row = m0 + wm + mt * 16 + r + half * 8;  // token
        int col = n0 + wn + nt * 16 + l16;           // feature
        float v = acc[mt][nt][r] + bias[col];
        int b = row >> 11, s = row & (SEQ - 1), h = col >> 6, d = col & 63;
        size_t bh = (size_t)(b * NHEAD + h);
        if (wsel == 0)
          qout[(bh * SEQ + s) * HDIM + d] = f2bf(v * 0.125f);  // fold 1/sqrt(64)
        else if (wsel == 1)
          kout[(bh * SEQ + s) * HDIM + d] = f2bf(v);
        else
          vtout[(bh * HDIM + d) * SEQ + s] = f2bf(v);
      }
}

// ---------- kernel 2: flash attention over (b, h, 128-query block) ----------

#define PITCH2 72   // 144B pitch: 16B-aligned, conflict-free for 16-row frag reads

__global__ __launch_bounds__(256, 1) void attn(
    const unsigned short* __restrict__ Q, const unsigned short* __restrict__ K,
    const unsigned short* __restrict__ Vt, unsigned short* __restrict__ Aout) {
  __shared__ unsigned short lK[2][64 * PITCH2];
  __shared__ unsigned short lV[2][64 * PITCH2];
  __shared__ unsigned short lP[8 * 16 * PITCH2];  // per-wave P scratch

  const int qb = blockIdx.x, h = blockIdx.y, b = blockIdx.z;
  const int tid = threadIdx.x;
  const int wave = tid >> 5, lane = tid & 31, half = lane >> 4, l16 = lane & 15;
  const size_t bh = (size_t)(b * NHEAD + h);
  const int q0 = qb * 128 + wave * 16;
  unsigned short* myP = &lP[wave * 16 * PITCH2];

  // this thread's two 16B staging chunks (64 rows x 8 chunks = 512 chunks)
  const int r0 = tid >> 3, o0 = (tid & 7) * 8;
  const int r1 = (tid + 256) >> 3, o1 = ((tid + 256) & 7) * 8;
  const unsigned short* Kb = K + bh * SEQ * HDIM;   // [key][d]
  const unsigned short* Vb = Vt + bh * HDIM * SEQ;  // [d][key]

  // Q fragments (16 queries x 64 dims), loaded once in A-layout
  bf16x16 qf[2];
  {
    const unsigned short* qp = Q + (bh * SEQ + q0 + l16) * HDIM;
#pragma unroll
    for (int ks = 0; ks < 2; ks++) {
      const unsigned short* p = qp + ks * 32 + half * 8;
      qf[ks] = mk_frag(*(const uint4*)p, *(const uint4*)(p + 16));
    }
  }

  f32x8 o[4];
  float m[8], l[8];
#pragma unroll
  for (int nt = 0; nt < 4; nt++)
#pragma unroll
    for (int r = 0; r < 8; r++) o[nt][r] = 0.0f;
#pragma unroll
  for (int r = 0; r < 8; r++) { m[r] = -INFINITY; l[r] = 0.0f; }

  // prologue: async-stage key-block 0 into buffer 0
  g2l_b128(&lK[0][r0 * PITCH2 + o0], &Kb[(size_t)r0 * HDIM + o0]);
  g2l_b128(&lK[0][r1 * PITCH2 + o1], &Kb[(size_t)r1 * HDIM + o1]);
  g2l_b128(&lV[0][r0 * PITCH2 + o0], &Vb[(size_t)r0 * SEQ + o0]);
  g2l_b128(&lV[0][r1 * PITCH2 + o1], &Vb[(size_t)r1 * SEQ + o1]);
  wait_async0();
  __syncthreads();

  int buf = 0;
  for (int kb = 0; kb < SEQ; kb += 64) {
    const int nb = buf ^ 1;
    if (kb + 64 < SEQ) {  // uniform: async-prefetch next K/V tiles
      const int kn = kb + 64;
      g2l_b128(&lK[nb][r0 * PITCH2 + o0], &Kb[(size_t)(kn + r0) * HDIM + o0]);
      g2l_b128(&lK[nb][r1 * PITCH2 + o1], &Kb[(size_t)(kn + r1) * HDIM + o1]);
      g2l_b128(&lV[nb][r0 * PITCH2 + o0], &Vb[(size_t)r0 * SEQ + kn + o0]);
      g2l_b128(&lV[nb][r1 * PITCH2 + o1], &Vb[(size_t)r1 * SEQ + kn + o1]);
    }

    // S = Q @ K^T  (16 x 64 logits; scale pre-folded into Q)
    f32x8 s[4];
#pragma unroll
    for (int nt = 0; nt < 4; nt++)
#pragma unroll
      for (int r = 0; r < 8; r++) s[nt][r] = 0.0f;
#pragma unroll
    for (int ks = 0; ks < 2; ks++)
#pragma unroll
      for (int nt = 0; nt < 4; nt++) {
        const unsigned short* p =
            &lK[buf][(nt * 16 + l16) * PITCH2 + ks * 32 + half * 8];
        bf16x16 kf = mk_frag(*(const uint4*)p, *(const uint4*)(p + 16));
        s[nt] = wmma_bf16(qf[ks], kf, s[nt]);
      }

    // online softmax (row = r + half*8, spread over a 16-lane half)
    float mn[8], al[8];
#pragma unroll
    for (int r = 0; r < 8; r++) {
      float mx = fmaxf(fmaxf(s[0][r], s[1][r]), fmaxf(s[2][r], s[3][r]));
#pragma unroll
      for (int d = 1; d < 16; d <<= 1) mx = fmaxf(mx, __shfl_xor(mx, d, 32));
      mn[r] = fmaxf(m[r], mx);
      al[r] = __expf(m[r] - mn[r]);
      m[r] = mn[r];
    }
#pragma unroll
    for (int r = 0; r < 8; r++) {
      float sum = 0.0f;
#pragma unroll
      for (int nt = 0; nt < 4; nt++) {
        float p = __expf(s[nt][r] - mn[r]);
        s[nt][r] = p;
        sum += p;
      }
#pragma unroll
      for (int d = 1; d < 16; d <<= 1) sum += __shfl_xor(sum, d, 32);
      l[r] = l[r] * al[r] + sum;
    }
#pragma unroll
    for (int nt = 0; nt < 4; nt++)
#pragma unroll
      for (int r = 0; r < 8; r++) o[nt][r] *= al[r];

    // C-layout -> A-layout via wave-private LDS round trip
#pragma unroll
    for (int nt = 0; nt < 4; nt++)
#pragma unroll
      for (int r = 0; r < 8; r++)
        myP[(r + half * 8) * PITCH2 + nt * 16 + l16] = f2bf(s[nt][r]);
    asm volatile("s_wait_dscnt 0" ::: "memory");  // DS ops in-order per wave

    // O += P @ V
#pragma unroll
    for (int ks = 0; ks < 2; ks++) {
      const unsigned short* pa = &myP[l16 * PITCH2 + ks * 32 + half * 8];
      bf16x16 pf = mk_frag(*(const uint4*)pa, *(const uint4*)(pa + 16));
#pragma unroll
      for (int nt = 0; nt < 4; nt++) {
        const unsigned short* pb =
            &lV[buf][(nt * 16 + l16) * PITCH2 + ks * 32 + half * 8];
        bf16x16 vf = mk_frag(*(const uint4*)pb, *(const uint4*)(pb + 16));
        o[nt] = wmma_bf16(pf, vf, o[nt]);
      }
    }

    wait_async0();
    __syncthreads();
    buf = nb;
  }

  // normalize and write attn-out in [b, s, h*64+d] layout for the Wo GEMM
#pragma unroll
  for (int r = 0; r < 8; r++) {
    float inv = 1.0f / l[r];
    int row = q0 + r + half * 8;
#pragma unroll
    for (int nt = 0; nt < 4; nt++)
      Aout[((size_t)b * SEQ + row) * EMBED + h * HDIM + nt * 16 + l16] =
          f2bf(o[nt][r] * inv);
  }
}

// ---------- kernel 3: output projection (fp32 result) ----------

__global__ __launch_bounds__(256, 1) void out_gemm(
    const unsigned short* __restrict__ Abf, const unsigned short* __restrict__ Wo,
    const float* __restrict__ bo, float* __restrict__ out) {
  __shared__ unsigned short lA[2][128 * PITCH1];
  __shared__ unsigned short lB[2][128 * PITCH1];

  const int m0 = blockIdx.y * 128, n0 = blockIdx.x * 128;
  const int tid = threadIdx.x;
  const int wave = tid >> 5, lane = tid & 31, half = lane >> 4, l16 = lane & 15;
  const int wm = (wave & 1) * 64, wn = (wave >> 1) * 32;
  const int r0 = tid >> 2, o0 = (tid & 3) * 8;
  const int r1 = (tid + 256) >> 2, o1 = ((tid + 256) & 3) * 8;

  f32x8 acc[4][2];
#pragma unroll
  for (int i = 0; i < 4; i++)
#pragma unroll
    for (int j = 0; j < 2; j++)
#pragma unroll
      for (int r = 0; r < 8; r++) acc[i][j][r] = 0.0f;

  g2l_b128(&lA[0][r0 * PITCH1 + o0], &Abf[(size_t)(m0 + r0) * EMBED + o0]);
  g2l_b128(&lA[0][r1 * PITCH1 + o1], &Abf[(size_t)(m0 + r1) * EMBED + o1]);
  g2l_b128(&lB[0][r0 * PITCH1 + o0], &Wo[(size_t)(n0 + r0) * EMBED + o0]);
  g2l_b128(&lB[0][r1 * PITCH1 + o1], &Wo[(size_t)(n0 + r1) * EMBED + o1]);
  wait_async0();
  __syncthreads();

  int buf = 0;
  for (int k0 = 0; k0 < EMBED; k0 += 32) {
    const int nb = buf ^ 1;
    if (k0 + 32 < EMBED) {
      const int kk = k0 + 32;
      g2l_b128(&lA[nb][r0 * PITCH1 + o0], &Abf[(size_t)(m0 + r0) * EMBED + kk + o0]);
      g2l_b128(&lA[nb][r1 * PITCH1 + o1], &Abf[(size_t)(m0 + r1) * EMBED + kk + o1]);
      g2l_b128(&lB[nb][r0 * PITCH1 + o0], &Wo[(size_t)(n0 + r0) * EMBED + kk + o0]);
      g2l_b128(&lB[nb][r1 * PITCH1 + o1], &Wo[(size_t)(n0 + r1) * EMBED + kk + o1]);
    }

    bf16x16 bfr[2];
#pragma unroll
    for (int nt = 0; nt < 2; nt++) {
      const unsigned short* p = &lB[buf][(wn + nt * 16 + l16) * PITCH1 + half * 8];
      bfr[nt] = mk_frag(*(const uint4*)p, *(const uint4*)(p + 16));
    }
#pragma unroll
    for (int mt = 0; mt < 4; mt++) {
      const unsigned short* p = &lA[buf][(wm + mt * 16 + l16) * PITCH1 + half * 8];
      bf16x16 af = mk_frag(*(const uint4*)p, *(const uint4*)(p + 16));
#pragma unroll
      for (int nt = 0; nt < 2; nt++) acc[mt][nt] = wmma_bf16(af, bfr[nt], acc[mt][nt]);
    }

    wait_async0();
    __syncthreads();
    buf = nb;
  }

#pragma unroll
  for (int mt = 0; mt < 4; mt++)
#pragma unroll
    for (int nt = 0; nt < 2; nt++)
#pragma unroll
      for (int r = 0; r < 8; r++) {
        int row = m0 + wm + mt * 16 + r + half * 8;
        int col = n0 + wn + nt * 16 + l16;
        out[(size_t)row * EMBED + col] = acc[mt][nt][r] + bo[col];
      }
}

// ---------- launch ----------

extern "C" void kernel_launch(void* const* d_in, const int* in_sizes, int n_in,
                              void* d_out, int out_size, void* d_ws, size_t ws_size,
                              hipStream_t stream) {
  const float* x  = (const float*)d_in[0];
  const float* Wq = (const float*)d_in[1];
  const float* bq = (const float*)d_in[2];
  const float* Wk = (const float*)d_in[3];
  const float* bk = (const float*)d_in[4];
  const float* Wv = (const float*)d_in[5];
  const float* bv = (const float*)d_in[6];
  const float* Wo = (const float*)d_in[7];
  const float* bo = (const float*)d_in[8];

  // workspace layout (bf16 halves), 88 MB total
  const size_t XBF = (size_t)TOK * EMBED;           // 8M elem
  const size_t WBF = (size_t)4 * EMBED * EMBED;     // 4M elem
  char* ws = (char*)d_ws;
  unsigned short* xbf = (unsigned short*)ws;                       ws += XBF * 2;
  unsigned short* wbf = (unsigned short*)ws;                       ws += WBF * 2;
  unsigned short* qbf = (unsigned short*)ws;                       ws += XBF * 2;
  unsigned short* kbf = (unsigned short*)ws;                       ws += XBF * 2;
  unsigned short* vtb = (unsigned short*)ws;                       ws += XBF * 2;
  unsigned short* abf = (unsigned short*)ws;

  cvt_x<<<dim3((TOK * EMBED / 4) / 256), dim3(256), 0, stream>>>(x, xbf);
  cvt_w<<<dim3(EMBED / 32, EMBED / 32, 4), dim3(32, 8), 0, stream>>>(Wq, Wk, Wv, Wo, wbf);
  qkv_gemm<<<dim3(EMBED / 128, TOK / 128, 3), dim3(256), 0, stream>>>(
      xbf, wbf, bq, bk, bv, qbf, kbf, vtb);
  attn<<<dim3(SEQ / 128, NHEAD, BATCH), dim3(256), 0, stream>>>(qbf, kbf, vtb, abf);
  out_gemm<<<dim3(EMBED / 128, TOK / 128), dim3(256), 0, stream>>>(
      abf, wbf + (size_t)3 * EMBED * EMBED, bo, (float*)d_out);
}
